// BinaryLinearLayer_41944650612857
// MI455X (gfx1250) — 8-bit matrix operand order repaired and checked by reading, compiled, not run on hardware
//
#include <hip/hip_runtime.h>


typedef int   v16i __attribute__((ext_vector_type(16)));
typedef float v8f  __attribute__((ext_vector_type(8)));
typedef int   g4i  __attribute__((vector_size(16)));

#define MDIM 8192
#define NDIM 4096
#define KDIM 4096
#define BM 128
#define BN 128
#define BK 128
#define NKT (KDIM / BK)

#if defined(__has_builtin)
#if __has_builtin(__builtin_amdgcn_global_load_async_to_lds_b128)
#define HAVE_ASYNC_LDS 1
#endif
#endif
#ifndef HAVE_ASYNC_LDS
#define HAVE_ASYNC_LDS 0
#endif


__device__ __forceinline__ unsigned int pack4_fp8pm1(float4 f) {
    unsigned int b0 = (f.x >= 0.0f) ? 0x38u : 0xB8u;
    unsigned int b1 = (f.y >= 0.0f) ? 0x38u : 0xB8u;
    unsigned int b2 = (f.z >= 0.0f) ? 0x38u : 0xB8u;
    unsigned int b3 = (f.w >= 0.0f) ? 0x38u : 0xB8u;
    return b0 | (b1 << 8) | (b2 << 16) | (b3 << 24);
}

__device__ __forceinline__ void g2lds64(const unsigned char* g, unsigned char* l) {
#if HAVE_ASYNC_LDS
    auto gp = (__attribute__((address_space(1))) g4i*)(void*)g;
    auto lp = (__attribute__((address_space(3))) g4i*)(void*)l;
    __builtin_amdgcn_global_load_async_to_lds_b128(gp, lp, 0, 0);
    __builtin_amdgcn_global_load_async_to_lds_b128(gp, lp, 16, 0);
    __builtin_amdgcn_global_load_async_to_lds_b128(gp, lp, 32, 0);
    __builtin_amdgcn_global_load_async_to_lds_b128(gp, lp, 48, 0);
#else
    const uint4* s = (const uint4*)g;
    uint4* d = (uint4*)l;
#pragma unroll
    for (int i = 0; i < 4; i++) d[i] = s[i];
#endif
}

template <int N>
__device__ __forceinline__ void wait_async() {
#if HAVE_ASYNC_LDS
#if defined(__has_builtin) && __has_builtin(__builtin_amdgcn_s_wait_asynccnt)
    __builtin_amdgcn_s_wait_asynccnt(N);
#else
    asm volatile("s_wait_asynccnt %0" ::"i"(N) : "memory");
#endif
#endif
}

__global__ __launch_bounds__(256) void binarize_x(const float* __restrict__ x,
                                                  uint4* __restrict__ xb,
                                                  int n16) {
    int i = blockIdx.x * 256 + threadIdx.x;
    if (i >= n16) return;
    const float4* s = (const float4*)x + (size_t)i * 4;
    unsigned int w0 = pack4_fp8pm1(s[0]);
    unsigned int w1 = pack4_fp8pm1(s[1]);
    unsigned int w2 = pack4_fp8pm1(s[2]);
    unsigned int w3 = pack4_fp8pm1(s[3]);
    xb[i] = make_uint4(w0, w1, w2, w3);
}

__global__ __launch_bounds__(256) void binarize_wT(const float* __restrict__ w,
                                                   unsigned char* __restrict__ wT) {
    __shared__ unsigned char tile[64 * 80];
    const int k0  = blockIdx.y * 64;
    const int nn0 = blockIdx.x * 64;
    const int t = threadIdx.x;
    const int r = t >> 2;
    const int c = (t & 3) * 16;

    const float4* src = (const float4*)(w + (size_t)(k0 + r) * NDIM + nn0 + c);
    unsigned int p0 = pack4_fp8pm1(src[0]);
    unsigned int p1 = pack4_fp8pm1(src[1]);
    unsigned int p2 = pack4_fp8pm1(src[2]);
    unsigned int p3 = pack4_fp8pm1(src[3]);
    *(uint4*)(tile + r * 80 + c) = make_uint4(p0, p1, p2, p3);
    __syncthreads();

    const int n  = r;
    const int kg = c;
    unsigned int q[4];
#pragma unroll
    for (int d = 0; d < 4; d++) {
        unsigned int v = 0;
#pragma unroll
        for (int j = 0; j < 4; j++)
            v |= (unsigned int)tile[(kg + d * 4 + j) * 80 + n] << (8 * j);
        q[d] = v;
    }
    *(uint4*)(wT + (size_t)(nn0 + n) * KDIM + k0 + kg) = make_uint4(q[0], q[1], q[2], q[3]);
}

__global__ __launch_bounds__(256) void bgemm_fp8(const unsigned char* __restrict__ Ab,
                                                 const unsigned char* __restrict__ BTb,
                                                 float* __restrict__ out) {
    __shared__ unsigned char As[2][BM * BK];
    __shared__ unsigned char Bs[2][BN * BK];

    const int tid  = threadIdx.x;
    const int lane = tid & 31;
    const int wave = tid >> 5;
    const int wm   = wave & 3;
    const int wn   = wave >> 2;
    const int h    = lane >> 4;
    const int l16  = lane & 15;

    const int m0 = blockIdx.y * BM;
    const int n0 = blockIdx.x * BN;

    const int gr = tid >> 1;
    const int gc = (tid & 1) * 64;
    const unsigned char* gA = Ab  + (size_t)(m0 + gr) * KDIM + gc;
    const unsigned char* gB = BTb + (size_t)(n0 + gr) * KDIM + gc;
    const int lofs = gr * BK + gc;

    v8f acc[2][4] = {};

    g2lds64(gA, &As[0][lofs]);
    g2lds64(gB, &Bs[0][lofs]);

    for (int it = 0; it < NKT; ++it) {
        const int p = it & 1;

        if (it + 1 < NKT) {
            const int kn = (it + 1) * BK;
            g2lds64(gA + kn, &As[p ^ 1][lofs]);
            g2lds64(gB + kn, &Bs[p ^ 1][lofs]);
            wait_async<8>();
        } else {
            wait_async<0>();
        }
        __syncthreads();

        v16i afr[2];
#pragma unroll
        for (int i = 0; i < 2; i++) {
            const unsigned char* base = &As[p][(wm * 32 + i * 16 + l16) * BK + 8 * h];
#pragma unroll
            for (int j = 0; j < 8; j++) {
                uint2 d = *(const uint2*)(base + 16 * j);
                afr[i][2 * j]     = (int)d.x;
                afr[i][2 * j + 1] = (int)d.y;
            }
        }

#pragma unroll
        for (int j = 0; j < 4; j++) {
            v16i bfr;
            const unsigned char* base = &Bs[p][(wn * 64 + j * 16 + l16) * BK + 8 * h];
#pragma unroll
            for (int q = 0; q < 8; q++) {
                uint2 d = *(const uint2*)(base + 16 * q);
                bfr[2 * q]     = (int)d.x;
                bfr[2 * q + 1] = (int)d.y;
            }
            acc[0][j] = __builtin_amdgcn_wmma_f32_16x16x128_fp8_fp8(
                afr[0], bfr, (short)0, acc[0][j], false, false);
            acc[1][j] = __builtin_amdgcn_wmma_f32_16x16x128_fp8_fp8(
                afr[1], bfr, (short)0, acc[1][j], false, false);
        }
        __syncthreads();
    }

#pragma unroll
    for (int i = 0; i < 2; i++) {
#pragma unroll
        for (int j = 0; j < 4; j++) {
            const int mrow = m0 + wm * 32 + i * 16 + 8 * h;
            const int ncol = n0 + wn * 64 + j * 16 + l16;
#pragma unroll
            for (int v = 0; v < 8; v++)
                out[(size_t)(mrow + v) * NDIM + ncol] = acc[i][j][v];
        }
    }
}

extern "C" void kernel_launch(void* const* d_in, const int* in_sizes, int n_in,
                              void* d_out, int out_size, void* d_ws, size_t ws_size,
                              hipStream_t stream) {
    const float* x = (const float*)d_in[0];
    const float* w = (const float*)d_in[1];
    float* out     = (float*)d_out;

    unsigned char* xb = (unsigned char*)d_ws;
    unsigned char* wT = xb + (size_t)MDIM * KDIM;

    const int n16 = (MDIM * KDIM) / 16;
    binarize_x<<<n16 / 256, 256, 0, stream>>>(x, (uint4*)xb, n16);
    binarize_wT<<<dim3(NDIM / 64, KDIM / 64), 256, 0, stream>>>(w, wT);
    bgemm_fp8<<<dim3(NDIM / BN, MDIM / BM), 256, 0, stream>>>(xb, wT, out);
}
